// GlobalAttention_41704132444718
// MI455X (gfx1250) — matrix operand order repaired and checked by reading, not compiled, not run on hardware
//
#include <hip/hip_runtime.h>
#include <hip/hip_bf16.h>

// GlobalAttention for MI455X (gfx1250, wave32).
// Projections via v_wmma_f32_16x16x32_bf16 (bf16 A/B, f32 accumulate);
// B-fragments fed from a transposed, padded LDS image of W so each lane
// loads its 32-byte fragment with two ds_load_b128 (no scalar u16 gathers).

#define Bn 64
#define Nn 512
#define Dn 128
#define NEG_INF_F (-1e9f)
#define WSTRIDE 136   // bf16 elements per transposed-W row: 128 + 8 pad
                      // (row stride 272 B -> lane bank = 4*lane mod 64, no conflicts;
                      //  272 % 16 == 0 keeps every fragment 16B-aligned for b128)

typedef __bf16 bf16_t;
typedef __attribute__((ext_vector_type(2)))  __bf16 v2bf;
typedef __attribute__((ext_vector_type(16))) __bf16 v16bf;
typedef __attribute__((ext_vector_type(8)))  float  v8f;

// ---------------------------------------------------------------------------
// Kernel 1: per (batch, 128-row slab): q/k projection tiles via WMMA.
// Each wave owns one 16-row M-tile and sweeps 8 N-tiles for both weights.
//   - writes mk = mask*k  to workspace             [B,N,D] f32
//   - writes selfdot[b,n] = mk[b,n] . mq[b,n]      [B,N]   f32 (shuffle tree)
//   - writes per-block partial q_sum (sum_n mq)    [B*4,D] f32 (no atomics)
// ---------------------------------------------------------------------------
__global__ __launch_bounds__(256) void ga_proj_wmma(
    const float* __restrict__ A,    const float* __restrict__ mask,
    const float* __restrict__ Wq,   const float* __restrict__ bqv,
    const float* __restrict__ Wk,   const float* __restrict__ bkv,
    float* __restrict__ mkbuf, float* __restrict__ selfdot,
    float* __restrict__ qsum_part)
{
    // W transposed: sWT[w][n * WSTRIDE + k], bf16.  2 * 128*136*2 B = 68 KB.
    __shared__ __attribute__((aligned(16))) bf16_t sWT[2][128 * WSTRIDE];
    __shared__ float  sMask[128];          // slab's mask values
    __shared__ float  sQ[8][128];          // per-wave q_sum partials

    const int tid  = threadIdx.x;
    const int lane = tid & 31;
    const int wave = tid >> 5;
    const int b    = blockIdx.x >> 2;      // batch
    const int slab = blockIdx.x & 3;       // 128-row slab within batch
    const int m0   = slab * 128;

    // Stage W^T as bf16: process K-pairs, packed v2bf store (ds_store_b32).
    // Global reads coalesced over n; LDS writes land on distinct banks.
    for (int i = tid; i < 64 * 128; i += 256) {
        const int kp = (i >> 7) * 2;       // even k
        const int n  = i & 127;
        v2bf pq, pk;
        pq[0] = (bf16_t)Wq[kp * 128 + n];
        pq[1] = (bf16_t)Wq[(kp + 1) * 128 + n];
        pk[0] = (bf16_t)Wk[kp * 128 + n];
        pk[1] = (bf16_t)Wk[(kp + 1) * 128 + n];
        *(v2bf*)&sWT[0][n * WSTRIDE + kp] = pq;
        *(v2bf*)&sWT[1][n * WSTRIDE + kp] = pk;
    }
    if (tid < 128) sMask[tid] = mask[b * Nn + m0 + tid];
    __syncthreads();

    const int mt   = m0 + wave * 16;       // this wave's M-tile base row
    const int half = lane >> 4;            // 0: lanes 0-15, 1: lanes 16-31
    const int l15  = lane & 15;

    // ---- A fragments: 16x32 bf16 per K-chunk, contiguous k order (same as B):
    //   lanes 0-15  : M=l15, K = 0..15
    //   lanes 16-31 : M=l15, K = 16..31
    v16bf afr[4];
    {
        const float* arow = A + (size_t)(b * Nn + mt + l15) * Dn;
        const int khalf = half * 16;
        for (int kk = 0; kk < 4; ++kk) {
            v16bf a;
#pragma unroll
            for (int i = 0; i < 8; ++i) {
                a[i]     = (bf16_t)arow[kk * 32 + khalf + i];
                a[8 + i] = (bf16_t)arow[kk * 32 + khalf + 8 + i];
            }
            afr[kk] = a;
        }
    }

    float maskv[8];
#pragma unroll
    for (int r = 0; r < 8; ++r) maskv[r] = sMask[wave * 16 + half * 8 + r];

    float sd[8];
#pragma unroll
    for (int r = 0; r < 8; ++r) sd[r] = 0.f;

    for (int nt = 0; nt < 8; ++nt) {
        const int ncol = nt * 16 + l15;

        // ---- B fragments: 32x16 bf16, contiguous k order (same as A):
        //   lanes 0-15 hold K=0..15 (col=l15), lanes 16-31 hold K=16..31.
        // With W^T in LDS these 16 values are 32 contiguous, 16B-aligned bytes.
        const int kbase = half * 16;
        const bf16_t* wq_ptr = &sWT[0][ncol * WSTRIDE + kbase];
        const bf16_t* wk_ptr = &sWT[1][ncol * WSTRIDE + kbase];

        v8f cq = {}; v8f ck = {};
#pragma unroll
        for (int kk = 0; kk < 4; ++kk) {
            const v16bf bfq = *(const v16bf*)(wq_ptr + kk * 32);
            const v16bf bfk = *(const v16bf*)(wk_ptr + kk * 32);
            cq = __builtin_amdgcn_wmma_f32_16x16x32_bf16(
                     false, afr[kk], false, bfq, (short)0, cq, false, false);
            ck = __builtin_amdgcn_wmma_f32_16x16x32_bf16(
                     false, afr[kk], false, bfk, (short)0, ck, false, false);
        }

        const float biasq = bqv[ncol];
        const float biask = bkv[ncol];

        // C/D layout: VGPR r -> row (half*8 + r), col = ncol.
        float qn = 0.f;
#pragma unroll
        for (int r = 0; r < 8; ++r) {
            const int row = mt + half * 8 + r;
            const float mqv  = maskv[r] * (cq[r] + biasq);
            const float mkv  = maskv[r] * (ck[r] + biask);
            mkbuf[(size_t)(b * Nn + row) * Dn + ncol] = mkv;
            sd[r] += mkv * mqv;          // self-dot partial (this dim)
            qn    += mqv;                // q_sum partial over 8 rows
        }
        // combine the two row-halves (lanes l and l^16 share ncol)
        qn += __shfl_xor(qn, 16, 32);
        if (half == 0) sQ[wave][nt * 16 + l15] = qn;
    }

    // selfdot: reduce over the 16 lanes of each half (16 dims x 8 nt = 128)
#pragma unroll
    for (int r = 0; r < 8; ++r) {
        float v = sd[r];
        v += __shfl_xor(v, 1, 32);
        v += __shfl_xor(v, 2, 32);
        v += __shfl_xor(v, 4, 32);
        v += __shfl_xor(v, 8, 32);
        if (l15 == 0) selfdot[b * Nn + mt + half * 8 + r] = v;
    }

    __syncthreads();
    if (tid < 128) {
        float acc = 0.f;
        for (int w = 0; w < 8; ++w) acc += sQ[w][tid];   // fixed order
        qsum_part[(size_t)blockIdx.x * 128 + tid] = acc;
    }
}

// ---------------------------------------------------------------------------
// Kernel 2: one block per batch. agg -> L2 normalize -> masked softmax ->
// attn output and context = sum_n attn[n]*mk[n,:]. All reductions are
// fixed-shape trees (deterministic).
// ---------------------------------------------------------------------------
__global__ __launch_bounds__(256) void ga_finalize(
    const float* __restrict__ mask,
    const float* __restrict__ mkbuf, const float* __restrict__ selfdot,
    const float* __restrict__ qsum_part,
    float* __restrict__ attn_out, float* __restrict__ ctx_out)
{
    __shared__ float sQsum[128];
    __shared__ float sAgg[512];
    __shared__ float sAttn[512];
    __shared__ float sRed[256];

    const int b    = blockIdx.x;
    const int tid  = threadIdx.x;
    const int lane = tid & 31;
    const int wave = tid >> 5;

    if (tid < 128) {
        float a = 0.f;
        for (int s = 0; s < 4; ++s)
            a += qsum_part[(size_t)(b * 4 + s) * 128 + tid];
        sQsum[tid] = a;
    }
    __syncthreads();

    // agg[n] = mask[n] * (mk[n].q_sum - selfdot[n]); one wave per row.
    for (int i = 0; i < 64; ++i) {
        const int n = wave * 64 + i;
        const float* mkrow = mkbuf + (size_t)(b * Nn + n) * Dn;
        float dot = 0.f;
#pragma unroll
        for (int j = 0; j < 4; ++j)
            dot += mkrow[lane + 32 * j] * sQsum[lane + 32 * j];
        dot += __shfl_xor(dot, 1, 32);
        dot += __shfl_xor(dot, 2, 32);
        dot += __shfl_xor(dot, 4, 32);
        dot += __shfl_xor(dot, 8, 32);
        dot += __shfl_xor(dot, 16, 32);
        const float ag = mask[b * Nn + n] * (dot - selfdot[b * Nn + n]);
        if (lane == 0) sAgg[n] = ag;
    }
    __syncthreads();

    // L2 norm over the 512 rows.
    float ss = sAgg[tid] * sAgg[tid] + sAgg[tid + 256] * sAgg[tid + 256];
    sRed[tid] = ss; __syncthreads();
    for (int off = 128; off > 0; off >>= 1) {
        if (tid < off) sRed[tid] += sRed[tid + off];
        __syncthreads();
    }
    const float nrm = sqrtf(sRed[0]);
    __syncthreads();

    const float mk0 = mask[b * Nn + tid];
    const float mk1 = mask[b * Nn + tid + 256];
    const float v0 = sAgg[tid] / nrm       + (1.f - mk0) * NEG_INF_F;
    const float v1 = sAgg[tid + 256] / nrm + (1.f - mk1) * NEG_INF_F;

    // max reduce
    sRed[tid] = fmaxf(v0, v1); __syncthreads();
    for (int off = 128; off > 0; off >>= 1) {
        if (tid < off) sRed[tid] = fmaxf(sRed[tid], sRed[tid + off]);
        __syncthreads();
    }
    const float vmax = sRed[0];
    __syncthreads();

    const float e0 = __expf(v0 - vmax);
    const float e1 = __expf(v1 - vmax);
    sRed[tid] = e0 + e1; __syncthreads();
    for (int off = 128; off > 0; off >>= 1) {
        if (tid < off) sRed[tid] += sRed[tid + off];
        __syncthreads();
    }
    const float Z = sRed[0];
    __syncthreads();

    const float a0 = e0 / Z, a1 = e1 / Z;
    attn_out[b * Nn + tid]       = a0;
    attn_out[b * Nn + tid + 256] = a1;
    sAttn[tid] = a0; sAttn[tid + 256] = a1;
    __syncthreads();

    // context[d] = sum_n attn[n] * mk[n,d]; 128 threads, coalesced over d.
    if (tid < 128) {
        float acc = 0.f;
        for (int n = 0; n < 512; ++n)
            acc += sAttn[n] * mkbuf[(size_t)(b * Nn + n) * Dn + tid];
        ctx_out[b * Dn + tid] = acc;
    }
}

// ---------------------------------------------------------------------------
extern "C" void kernel_launch(void* const* d_in, const int* in_sizes, int n_in,
                              void* d_out, int out_size, void* d_ws, size_t ws_size,
                              hipStream_t stream) {
    const float* A    = (const float*)d_in[0];   // [B,N,D]
    const float* mask = (const float*)d_in[1];   // [B,N,1]
    const float* Wq   = (const float*)d_in[2];   // [D,DIM]
    const float* bq   = (const float*)d_in[3];   // [DIM]
    const float* Wk   = (const float*)d_in[4];   // [D,DIM]
    const float* bk   = (const float*)d_in[5];   // [DIM]

    // workspace layout (f32): mk [B*N*D] | selfdot [B*N] | qsum_part [B*4*128]
    float* ws_mk  = (float*)d_ws;
    float* ws_sd  = ws_mk + (size_t)Bn * Nn * Dn;            // +4,194,304
    float* ws_qp  = ws_sd + (size_t)Bn * Nn;                 // +32,768
    // total: (4194304+32768+32768)*4 B ~= 16.3 MiB

    float* attn = (float*)d_out;                 // [B,N,1] = 32768
    float* ctx  = attn + (size_t)Bn * Nn;        // [B,DIM] =  8192

    ga_proj_wmma<<<Bn * 4, 256, 0, stream>>>(A, mask, Wq, bq, Wk, bk,
                                             ws_mk, ws_sd, ws_qp);
    ga_finalize<<<Bn, 256, 0, stream>>>(mask, ws_mk, ws_sd, ws_qp, attn, ctx);
}
